// GCN_Graph_3367254360556
// MI455X (gfx1250) — matrix operand order repaired and checked by reading, not compiled, not run on hardware
//
#include <hip/hip_runtime.h>
#include <hip/hip_bf16.h>
#include <stdint.h>

// ---------------- problem constants (match reference) ----------------
#define N_NODES 100000
#define N_EDGES 1600000
#define N_FEAT  128
#define N_HID   128
#define N_CLASS 8
#define N_GRAPH 256

typedef __attribute__((ext_vector_type(16))) _Float16 v16h;
typedef __attribute__((ext_vector_type(8)))  float    v8f;

__device__ __forceinline__ uint32_t packh(float lo, float hi) {
    union { _Float16 h[2]; uint32_t u; } p;
    p.h[0] = (_Float16)lo; p.h[1] = (_Float16)hi;
    return p.u;
}

// ---------------------------------------------------------------------
// K0: pack W1/W2 into the WMMA B-operand lane layout (f16), zero pooled/cnt
// B layout (16-bit, 32x16, wave32): VGPR j: lanes 0-15 hold K=2j,2j+1 (j<4) / 2j+8,2j+9 (j>=4) (N=lane);
// lanes 16-31 hold the same K plus 8 (N=lane-16).
// Bp1: [kb=4][nt=8][lane=32][j=8] u32 ; Bp2: [kb=4][lane=32][j=8] u32 (nt=0)
// ---------------------------------------------------------------------
__global__ void k_pack(const float* __restrict__ W1, const float* __restrict__ W2,
                       uint32_t* __restrict__ Bp1, uint32_t* __restrict__ Bp2,
                       float* __restrict__ pooled, float* __restrict__ cnt) {
    int idx = blockIdx.x * blockDim.x + threadIdx.x;
    if (idx < 8192) {                       // W1 pack
        int j    = idx & 7;
        int lane = (idx >> 3) & 31;
        int nt   = (idx >> 8) & 7;
        int kb   = idx >> 11;
        int n    = nt * 16 + (lane & 15);
        int k0   = kb * 32 + ((lane < 16) ? 0 : 8) + ((j < 4) ? 2 * j : 16 + 2 * (j - 4));
        Bp1[idx] = packh(W1[k0 * N_HID + n], W1[(k0 + 1) * N_HID + n]);
    } else if (idx < 8192 + 1024) {         // W2 pack (pad 8 -> 16 cols with 0)
        int i2   = idx - 8192;
        int j    = i2 & 7;
        int lane = (i2 >> 3) & 31;
        int kb   = i2 >> 8;
        int n    = lane & 15;
        int k0   = kb * 32 + ((lane < 16) ? 0 : 8) + ((j < 4) ? 2 * j : 16 + 2 * (j - 4));
        Bp2[i2] = (n < N_CLASS)
                    ? packh(W2[k0 * N_CLASS + n], W2[(k0 + 1) * N_CLASS + n])
                    : 0u;
    } else if (idx < 8192 + 1024 + N_GRAPH * N_CLASS) {
        pooled[idx - 9216] = 0.0f;
    } else if (idx < 8192 + 1024 + N_GRAPH * N_CLASS + N_GRAPH) {
        cnt[idx - 9216 - N_GRAPH * N_CLASS] = 0.0f;
    }
}

// ---------------- degree / normalization ----------------
__global__ void k_deg_init(float* __restrict__ deg) {
    int i = blockIdx.x * blockDim.x + threadIdx.x;
    if (i < N_NODES) deg[i] = 1.0f;              // self loop
}
__global__ void k_deg_edges(const int* __restrict__ edges, float* __restrict__ deg) {
    int e = blockIdx.x * blockDim.x + threadIdx.x;
    if (e < N_EDGES) atomicAdd(&deg[edges[N_EDGES + e]], 1.0f);
}
__global__ void k_dinv(float* __restrict__ deg) {
    int i = blockIdx.x * blockDim.x + threadIdx.x;
    if (i < N_NODES) deg[i] = rsqrtf(deg[i]);    // in-place -> dinv
}

// ---------------------------------------------------------------------
// GEMM1 (fused epilogue): xw1 = x @ W1  AND  acc1 = b1 + xw1*dinv^2
// (self-loop + bias seeded directly so the edge scatter can accumulate on top;
//  saves a full 51MB read + 51MB write pass vs. a separate seeding kernel).
// 256 thr = 8 waves; wave w owns rows [blk*128 + w*16, +16), full 128 cols.
// A built per K-step from f32 x (cvt to f16), ISA A-layout:
//   lane<16: K = {0..7, 16..23}; lane>=16: K = {8..15, 24..31} (koff=8).
// ---------------------------------------------------------------------
__global__ void k_gemm1(const float* __restrict__ x, const uint32_t* __restrict__ Bp1,
                        const float* __restrict__ dinv, const float* __restrict__ b1,
                        float* __restrict__ xw1, float* __restrict__ acc1) {
    const int lane    = threadIdx.x & 31;
    const int wave    = threadIdx.x >> 5;
    const int rowbase = blockIdx.x * 128 + wave * 16;
    const int rA      = min(rowbase + (lane & 15), N_NODES - 1);
    const int koff    = (lane < 16) ? 0 : 8;

    v8f c[8];
    #pragma unroll
    for (int nt = 0; nt < 8; ++nt) c[nt] = (v8f)0.0f;

    #pragma unroll
    for (int kb = 0; kb < 4; ++kb) {
        const float* xr = x + (size_t)rA * N_FEAT + kb * 32 + koff;
        v16h a;
        #pragma unroll
        for (int j = 0; j < 8; ++j) {
            int kk = (j < 4) ? 2 * j : 16 + 2 * (j - 4);
            float2 f = *(const float2*)(xr + kk);
            a[2 * j]     = (_Float16)f.x;
            a[2 * j + 1] = (_Float16)f.y;
        }
        #pragma unroll
        for (int nt = 0; nt < 8; ++nt) {
            v16h b = *(const v16h*)(Bp1 + (((kb * 8 + nt) * 32) + lane) * 8);
            c[nt] = __builtin_amdgcn_wmma_f32_16x16x32_f16(
                        false, a, false, b, (short)0, c[nt], false, false);
        }
    }

    // per-lane dinv^2 for this lane's 8 output rows (reused across 8 col tiles)
    float d2[8];
    #pragma unroll
    for (int r = 0; r < 8; ++r) {
        int row = rowbase + ((lane < 16) ? r : r + 8);
        float di = dinv[min(row, N_NODES - 1)];
        d2[r] = di * di;
    }
    // C/D layout: VGPR r -> M = r (lanes 0-15) / r+8 (lanes 16-31), N = lane&15
    #pragma unroll
    for (int nt = 0; nt < 8; ++nt) {
        int col = nt * 16 + (lane & 15);
        float bv = b1[col];
        #pragma unroll
        for (int r = 0; r < 8; ++r) {
            int row = rowbase + ((lane < 16) ? r : r + 8);
            if (row < N_NODES) {
                float val = c[nt][r];
                xw1 [(size_t)row * N_HID + col] = val;
                acc1[(size_t)row * N_HID + col] = bv + val * d2[r];
            }
        }
    }
}

// 32 lanes per edge: lane handles 4 consecutive feats (float4 gather, coalesced)
__global__ void k_scatter1(const int* __restrict__ edges, const float* __restrict__ dinv,
                           const float* __restrict__ xw1, float* __restrict__ acc1) {
    long long t = (long long)blockIdx.x * blockDim.x + threadIdx.x;
    if (t >= (long long)N_EDGES * 32) return;
    int e    = (int)(t >> 5);
    int lane = (int)(t & 31);
    int s = edges[e], d = edges[N_EDGES + e];
    float norm = dinv[s] * dinv[d];
    int f0 = lane * 4;
    float4 v = *(const float4*)(xw1 + (size_t)s * N_HID + f0);
    float* o = acc1 + (size_t)d * N_HID + f0;
    atomicAdd(o + 0, v.x * norm);
    atomicAdd(o + 1, v.y * norm);
    atomicAdd(o + 2, v.z * norm);
    atomicAdd(o + 3, v.w * norm);
}

// GEMM2 (fused): hw2 = relu(acc1) @ W2pad  AND  acc2 = b2 + hw2*dinv^2
__global__ void k_gemm2(const float* __restrict__ acc1, const uint32_t* __restrict__ Bp2,
                        const float* __restrict__ dinv, const float* __restrict__ b2,
                        float* __restrict__ hw2, float* __restrict__ acc2) {
    const int lane    = threadIdx.x & 31;
    const int wave    = threadIdx.x >> 5;
    const int rowbase = blockIdx.x * 128 + wave * 16;
    const int rA      = min(rowbase + (lane & 15), N_NODES - 1);
    const int koff    = (lane < 16) ? 0 : 8;

    v8f c = (v8f)0.0f;
    #pragma unroll
    for (int kb = 0; kb < 4; ++kb) {
        const float* xr = acc1 + (size_t)rA * N_HID + kb * 32 + koff;
        v16h a;
        #pragma unroll
        for (int j = 0; j < 8; ++j) {
            int kk = (j < 4) ? 2 * j : 16 + 2 * (j - 4);
            float2 f = *(const float2*)(xr + kk);
            a[2 * j]     = (_Float16)fmaxf(f.x, 0.0f);   // fused ReLU
            a[2 * j + 1] = (_Float16)fmaxf(f.y, 0.0f);
        }
        v16h b = *(const v16h*)(Bp2 + ((kb * 32) + lane) * 8);
        c = __builtin_amdgcn_wmma_f32_16x16x32_f16(
                false, a, false, b, (short)0, c, false, false);
    }

    int col = lane & 15;
    float bv = (col < N_CLASS) ? b2[col] : 0.0f;
    #pragma unroll
    for (int r = 0; r < 8; ++r) {
        int row = rowbase + ((lane < 16) ? r : r + 8);
        if (row < N_NODES) {
            float val = c[r];
            hw2[(size_t)row * 16 + col] = val;
            if (col < N_CLASS) {
                float di = dinv[row];
                acc2[(size_t)row * N_CLASS + col] = bv + val * di * di;
            }
        }
    }
}

// 8 lanes per edge
__global__ void k_scatter2(const int* __restrict__ edges, const float* __restrict__ dinv,
                           const float* __restrict__ hw2, float* __restrict__ acc2) {
    long long t = (long long)blockIdx.x * blockDim.x + threadIdx.x;
    if (t >= (long long)N_EDGES * 8) return;
    int e  = (int)(t >> 3);
    int cc = (int)(t & 7);
    int s = edges[e], d = edges[N_EDGES + e];
    float norm = dinv[s] * dinv[d];
    atomicAdd(&acc2[(size_t)d * N_CLASS + cc],
              hw2[(size_t)s * 16 + cc] * norm);
}

// mean-pool accumulation over graphs
__global__ void k_pool(const float* __restrict__ acc2, const int* __restrict__ batch,
                       float* __restrict__ pooled, float* __restrict__ cnt) {
    long long t = (long long)blockIdx.x * blockDim.x + threadIdx.x;
    if (t >= (long long)N_NODES * N_CLASS) return;
    int i = (int)(t >> 3), cc = (int)(t & 7);
    int g = batch[i];
    atomicAdd(&pooled[g * N_CLASS + cc], acc2[t]);
    if (cc == 0) atomicAdd(&cnt[g], 1.0f);
}

// finalize: mean + log_softmax, one thread per graph
__global__ void k_final(const float* __restrict__ pooled, const float* __restrict__ cnt,
                        float* __restrict__ out) {
    int g = blockIdx.x * blockDim.x + threadIdx.x;
    if (g >= N_GRAPH) return;
    float cn = fmaxf(cnt[g], 1.0f);
    float v[N_CLASS], m = -1e30f;
    #pragma unroll
    for (int c = 0; c < N_CLASS; ++c) { v[c] = pooled[g * N_CLASS + c] / cn; m = fmaxf(m, v[c]); }
    float s = 0.0f;
    #pragma unroll
    for (int c = 0; c < N_CLASS; ++c) s += expf(v[c] - m);
    float l = logf(s);
    #pragma unroll
    for (int c = 0; c < N_CLASS; ++c) out[g * N_CLASS + c] = v[c] - m - l;
}

// ---------------------------------------------------------------------
extern "C" void kernel_launch(void* const* d_in, const int* in_sizes, int n_in,
                              void* d_out, int out_size, void* d_ws, size_t ws_size,
                              hipStream_t stream) {
    const float* x     = (const float*)d_in[0];   // [N,128]
    const int*   edges = (const int*)  d_in[1];   // [2,E] flat
    const int*   batch = (const int*)  d_in[2];   // [N]
    const float* W1    = (const float*)d_in[3];   // [128,128]
    const float* b1    = (const float*)d_in[4];   // [128]
    const float* W2    = (const float*)d_in[5];   // [128,8]
    const float* b2    = (const float*)d_in[6];   // [8]
    float* out = (float*)d_out;                   // [256,8]

    // workspace carve-up (256B aligned)
    size_t off = 0;
    auto carve = [&](size_t bytes) { size_t o = off; off = (off + bytes + 255) & ~(size_t)255; return o; };
    char* ws = (char*)d_ws;
    float*    dinv   = (float*)   (ws + carve(sizeof(float)    * N_NODES));
    float*    xw1    = (float*)   (ws + carve(sizeof(float)    * (size_t)N_NODES * N_HID));
    float*    acc1   = (float*)   (ws + carve(sizeof(float)    * (size_t)N_NODES * N_HID));
    float*    hw2    = (float*)   (ws + carve(sizeof(float)    * (size_t)N_NODES * 16));
    float*    acc2   = (float*)   (ws + carve(sizeof(float)    * (size_t)N_NODES * N_CLASS));
    float*    pooled = (float*)   (ws + carve(sizeof(float)    * N_GRAPH * N_CLASS));
    float*    cnt    = (float*)   (ws + carve(sizeof(float)    * N_GRAPH));
    uint32_t* Bp1    = (uint32_t*)(ws + carve(sizeof(uint32_t) * 8192));
    uint32_t* Bp2    = (uint32_t*)(ws + carve(sizeof(uint32_t) * 1024));
    (void)ws_size; (void)n_in; (void)in_sizes; (void)out_size;

    const int T = 256;
    // weight packing + zero pool buffers
    k_pack<<<(8192 + 1024 + N_GRAPH * N_CLASS + N_GRAPH + T - 1) / T, T, 0, stream>>>(
        W1, W2, Bp1, Bp2, pooled, cnt);
    // degrees -> dinv (must precede GEMM1, whose epilogue consumes dinv)
    k_deg_init<<<(N_NODES + T - 1) / T, T, 0, stream>>>(dinv);
    k_deg_edges<<<(N_EDGES + T - 1) / T, T, 0, stream>>>(edges, dinv);
    k_dinv<<<(N_NODES + T - 1) / T, T, 0, stream>>>(dinv);
    // layer 1: WMMA GEMM with fused bias+self-loop seed, then edge scatter
    k_gemm1<<<(N_NODES + 127) / 128, T, 0, stream>>>(x, Bp1, dinv, b1, xw1, acc1);
    k_scatter1<<<(int)(((long long)N_EDGES * 32 + T - 1) / T), T, 0, stream>>>(edges, dinv, xw1, acc1);
    // layer 2: fused ReLU + WMMA GEMM + bias/self-loop seed, then edge scatter
    k_gemm2<<<(N_NODES + 127) / 128, T, 0, stream>>>(acc1, Bp2, dinv, b2, hw2, acc2);
    k_scatter2<<<(int)(((long long)N_EDGES * 8 + T - 1) / T), T, 0, stream>>>(edges, dinv, hw2, acc2);
    // pool + log_softmax
    k_pool<<<(int)(((long long)N_NODES * N_CLASS + T - 1) / T), T, 0, stream>>>(acc2, batch, pooled, cnt);
    k_final<<<1, 256, 0, stream>>>(pooled, cnt, out);
}
